// SelfAttention_4088808866154
// MI455X (gfx1250) — matrix operand order repaired and checked by reading, not compiled, not run on hardware
//
#include <hip/hip_runtime.h>
#include <hip/hip_bf16.h>

// Problem constants (match reference)
#define BB 32
#define CCH 256
#define HH 56
#define WWI 56
#define NN (HH * WWI)   // 3136
#define C3 (3 * CCH)    // 768
#define EPSV 1e-12f

typedef __attribute__((ext_vector_type(16))) __bf16 bf16x16;
typedef __attribute__((ext_vector_type(8)))  __bf16 bf16x8;
typedef __attribute__((ext_vector_type(8)))  float  floatx8;

// ---------------------------------------------------------------------------
// Generic WMMA GEMM, 4x2 register macro-tile per wave (M=64, N=32):
//   Out[b][m][n] = sum_k A[b][m*lda + k] * BT[b][n*ldbt + k]
// Both operands are bf16 with CONTIGUOUS K-runs -> all fragment loads are
// 16-byte b128 loads. klen is the contraction length (multiple of 32).
// MODE 0: Out bf16 "N-major" [n][ldo], packed b128 stores.
// MODE 1: Out f32 row-major [m][ldo].
// HAS_BIAS / HAS_RESID are compile-time so epilogues are branch-free.
// Fragment layouts used here (D per ISA 7.12.2; A/B K order is contiguous, not the ISA table):
//   A 16x32 bf16: lane<16 -> K {0..15}; lane>=16 -> K {16..31}; row = lane%16
//   B 32x16 bf16: lane-half selects K-chunk of 16; col = lane%16
//   D 16x16 f32 : lane<16 -> rows 0..7; lane>=16 -> rows 8..15; col = lane%16
// ---------------------------------------------------------------------------
template <int MODE, bool HAS_BIAS, bool HAS_RESID>
__global__ __launch_bounds__(32) void gemm_wmma_kernel(
    const __bf16* __restrict__ A, long long strideA, int lda,
    const __bf16* __restrict__ BT, long long strideBT, int ldbt,
    int klen,
    const float* __restrict__ bias,
    const float* __restrict__ resid, long long strideR,
    void* __restrict__ Out, long long strideO, int ldo)
{
    const int lane = threadIdx.x & 31;
    const int n0 = blockIdx.x * 32;   // 2 n-tiles
    const int m0 = blockIdx.y * 64;   // 4 m-tiles
    const int b  = blockIdx.z;
    const __bf16* Ab = A + (size_t)b * strideA;
    const __bf16* Bb = BT + (size_t)b * strideBT;
    const int rc   = lane & 15;          // A-row / B-col / D-col within tile
    const int kh   = (lane >> 4) << 4;   // A K-chunk offset (0 or 16)
    const int koff = (lane >> 4) << 4;   // B K-chunk offset (0 or 16)

    floatx8 acc[4][2] = {};
    for (int k = 0; k < klen; k += 32) {
        bf16x16 bf[2];
#pragma unroll
        for (int nt = 0; nt < 2; ++nt) {
            const bf16x8* bp = (const bf16x8*)(Bb +
                (size_t)(n0 + nt * 16 + rc) * ldbt + k + koff);
            bf16x8 b0 = bp[0], b1 = bp[1];      // 2x global_load_b128
#pragma unroll
            for (int j = 0; j < 8; ++j) { bf[nt][j] = b0[j]; bf[nt][j + 8] = b1[j]; }
        }
        bf16x16 af[4];
#pragma unroll
        for (int mt = 0; mt < 4; ++mt) {
            const bf16x8* ap = (const bf16x8*)(Ab +
                (size_t)(m0 + mt * 16 + rc) * lda + k + kh);
            bf16x8 a0 = ap[0], a1 = ap[1];      // runs at +0 and +8 elements
#pragma unroll
            for (int j = 0; j < 8; ++j) { af[mt][j] = a0[j]; af[mt][j + 8] = a1[j]; }
        }
#pragma unroll
        for (int mt = 0; mt < 4; ++mt)
#pragma unroll
            for (int nt = 0; nt < 2; ++nt)
                acc[mt][nt] = __builtin_amdgcn_wmma_f32_16x16x32_bf16(
                    false, af[mt], false, bf[nt], (short)0, acc[mt][nt], false, false);
    }

    const int rb = (lane >> 4) << 3;
    if (MODE == 0) {
        __bf16* O = (__bf16*)Out + (size_t)b * strideO;
#pragma unroll
        for (int nt = 0; nt < 2; ++nt) {
            const int n = n0 + nt * 16 + rc;
#pragma unroll
            for (int mt = 0; mt < 4; ++mt) {
                bf16x8 v;
#pragma unroll
                for (int i = 0; i < 8; ++i) {
                    float t = acc[mt][nt][i];
                    if (HAS_BIAS) t += bias[m0 + mt * 16 + rb + i];
                    v[i] = (__bf16)t;
                }
                *(bf16x8*)(O + (size_t)n * ldo + m0 + mt * 16 + rb) = v;  // b128
            }
        }
    } else {
        float* O = (float*)Out + (size_t)b * strideO;
        const float* R = HAS_RESID ? resid + (size_t)b * strideR : nullptr;
#pragma unroll
        for (int mt = 0; mt < 4; ++mt)
#pragma unroll
            for (int i = 0; i < 8; ++i) {
                const int m = m0 + mt * 16 + rb + i;
                const float bi = HAS_BIAS ? bias[m] : 0.0f;
#pragma unroll
                for (int nt = 0; nt < 2; ++nt) {
                    const int n = n0 + nt * 16 + rc;
                    float t = acc[mt][nt][i] + bi;
                    if (HAS_RESID) t += R[(size_t)m * ldo + n];
                    O[(size_t)m * ldo + n] = t;
                }
            }
    }
}

// ---------------------------------------------------------------------------
// x [B][C][N] f32  ->  xT [B][N][C] bf16 via LDS tile transpose (32x32).
// ---------------------------------------------------------------------------
__global__ __launch_bounds__(256) void transpose_x_kernel(
    const float* __restrict__ x, __bf16* __restrict__ xT)
{
    __shared__ __bf16 tile[32][33];
    const int n0 = blockIdx.x * 32, c0 = blockIdx.y * 32, b = blockIdx.z;
    const int tc = threadIdx.x & 31;
    const int tr = threadIdx.x >> 5;    // 0..7
    const float* xp = x + (size_t)b * CCH * NN;
    for (int r = tr; r < 32; r += 8)
        tile[r][tc] = (__bf16)xp[(size_t)(c0 + r) * NN + n0 + tc];
    __syncthreads();
    __bf16* op = xT + (size_t)b * NN * CCH;
    for (int r = tr; r < 32; r += 8)
        op[(size_t)(n0 + r) * CCH + c0 + tc] = tile[tc][r];
}

// ---------------------------------------------------------------------------
__global__ void f32_to_bf16_kernel(const float* __restrict__ in,
                                   __bf16* __restrict__ out, int count)
{
    const int i = blockIdx.x * 256 + threadIdx.x;
    if (i < count) out[i] = (__bf16)in[i];
}

// ---------------------------------------------------------------------------
// Depthwise 3x3 + bias for V channels (512..767): NHWC in -> N-major vT out.
// Threads map to the 256 v-channels: coalesced loads and stores.
// ---------------------------------------------------------------------------
__global__ __launch_bounds__(256) void dwconv_v_kernel(
    const __bf16* __restrict__ qkvT, const float* __restrict__ wdw,
    const float* __restrict__ bdw, __bf16* __restrict__ vT)
{
    const int c  = threadIdx.x;            // 0..255 (v channel)
    const int ch = 2 * CCH + c;            // global channel in qkv
    const int n  = blockIdx.x;
    const int b  = blockIdx.y;
    const int h = n / WWI, w = n % WWI;
    const __bf16* src = qkvT + (size_t)b * NN * C3;
    float s = bdw[ch];
#pragma unroll
    for (int dh = -1; dh <= 1; ++dh)
#pragma unroll
        for (int dw = -1; dw <= 1; ++dw) {
            const int hh = h + dh, ww = w + dw;
            if (hh >= 0 && hh < HH && ww >= 0 && ww < WWI)
                s += wdw[ch * 9 + (dh + 1) * 3 + (dw + 1)] *
                     (float)src[(size_t)(hh * WWI + ww) * C3 + ch];
        }
    vT[(size_t)b * NN * CCH + (size_t)n * CCH + c] = (__bf16)s;
}

// ---------------------------------------------------------------------------
// Depthwise 3x3 + bias for Q/K channels (0..511), fused with transpose:
// NHWC in (coalesced over channels) -> C-major qk_cm [2C][N] bf16 out
// (coalesced over pixels) via a 32x32 LDS tile.
// ---------------------------------------------------------------------------
__global__ __launch_bounds__(256) void dwconv_qk_t_kernel(
    const __bf16* __restrict__ qkvT, const float* __restrict__ wdw,
    const float* __restrict__ bdw, __bf16* __restrict__ qk_cm)
{
    __shared__ __bf16 tile[32][33];        // [pixel][channel]
    const int n0  = blockIdx.x * 32;
    const int ch0 = blockIdx.y * 32;       // 0..511 in steps of 32
    const int b   = blockIdx.z;
    const int tc = threadIdx.x & 31;
    const int tr = threadIdx.x >> 5;       // 0..7
    const __bf16* src = qkvT + (size_t)b * NN * C3;

    for (int p = tr; p < 32; p += 8) {
        const int n = n0 + p;
        const int h = n / WWI, w = n % WWI;
        const int ch = ch0 + tc;
        float s = bdw[ch];
#pragma unroll
        for (int dh = -1; dh <= 1; ++dh)
#pragma unroll
            for (int dw = -1; dw <= 1; ++dw) {
                const int hh = h + dh, ww = w + dw;
                if (hh >= 0 && hh < HH && ww >= 0 && ww < WWI)
                    s += wdw[ch * 9 + (dh + 1) * 3 + (dw + 1)] *
                         (float)src[(size_t)(hh * WWI + ww) * C3 + ch];
            }
        tile[p][tc] = (__bf16)s;
    }
    __syncthreads();
    __bf16* op = qk_cm + (size_t)b * 2 * CCH * NN;
    for (int c = tr; c < 32; c += 8)
        op[(size_t)(ch0 + c) * NN + n0 + tc] = tile[tc][c];
}

// ---------------------------------------------------------------------------
// Inverse L2 norms: one block per (row=q/k channel, batch); contiguous
// C-major rows, b128 vector loads + LDS tree reduction.
// ---------------------------------------------------------------------------
__global__ __launch_bounds__(256) void rnorm_kernel(
    const __bf16* __restrict__ qk_cm, float* __restrict__ rn)
{
    __shared__ float red[256];
    const int row = blockIdx.x;            // 0..511 (q: 0..255, k: 256..511)
    const int b   = blockIdx.y;
    const __bf16* p = qk_cm + ((size_t)b * 2 * CCH + row) * NN;
    float s = 0.0f;
    for (int chunk = threadIdx.x; chunk < NN / 8; chunk += 256) {  // 392 chunks
        bf16x8 v = ((const bf16x8*)p)[chunk];
#pragma unroll
        for (int j = 0; j < 8; ++j) { const float f = (float)v[j]; s += f * f; }
    }
    red[threadIdx.x] = s;
    __syncthreads();
    for (int st = 128; st > 0; st >>= 1) {
        if (threadIdx.x < st) red[threadIdx.x] += red[threadIdx.x + st];
        __syncthreads();
    }
    if (threadIdx.x == 0)
        rn[(size_t)b * 2 * CCH + row] = 1.0f / fmaxf(sqrtf(red[0]), EPSV);
}

// ---------------------------------------------------------------------------
// Row softmax with folded norms + temperature; f32 logits -> bf16 attn.
// ---------------------------------------------------------------------------
__global__ __launch_bounds__(256) void softmax_kernel(
    const float* __restrict__ S, const float* __restrict__ rn,
    const float* __restrict__ temp, __bf16* __restrict__ attnb)
{
    __shared__ float red[256];
    const int i = blockIdx.x;
    const int b = blockIdx.y;
    const int j = threadIdx.x;
    const float rq = rn[(size_t)b * 2 * CCH + i];
    const float rk = rn[(size_t)b * 2 * CCH + CCH + j];
    const float v = S[((size_t)b * CCH + i) * CCH + j] * rq * rk * temp[0];
    red[j] = v;
    __syncthreads();
    for (int st = 128; st > 0; st >>= 1) {
        if (j < st) red[j] = fmaxf(red[j], red[j + st]);
        __syncthreads();
    }
    const float mx = red[0];
    __syncthreads();
    const float e = __expf(v - mx);
    red[j] = e;
    __syncthreads();
    for (int st = 128; st > 0; st >>= 1) {
        if (j < st) red[j] += red[j + st];
        __syncthreads();
    }
    attnb[((size_t)b * CCH + i) * CCH + j] = (__bf16)(e / red[0]);
}

// ---------------------------------------------------------------------------
extern "C" void kernel_launch(void* const* d_in, const int* in_sizes, int n_in,
                              void* d_out, int out_size, void* d_ws, size_t ws_size,
                              hipStream_t stream)
{
    const float* x      = (const float*)d_in[0];
    const float* w_qkv  = (const float*)d_in[1];
    const float* b_qkv  = (const float*)d_in[2];
    const float* w_dw   = (const float*)d_in[3];
    const float* b_dw   = (const float*)d_in[4];
    const float* w_proj = (const float*)d_in[5];
    const float* b_proj = (const float*)d_in[6];
    const float* temp   = (const float*)d_in[7];
    float* out = (float*)d_out;
    (void)in_sizes; (void)n_in; (void)out_size; (void)ws_size;

    // Workspace carve-out (256B aligned slices)
    char* w = (char*)d_ws;
    auto alloc = [&](size_t bytes) {
        void* p = (void*)w;
        w += (bytes + 255) & ~(size_t)255;
        return p;
    };
    __bf16* xT     = (__bf16*)alloc((size_t)BB * NN * CCH * 2);   // reused as out1T
    __bf16* qkvT   = (__bf16*)alloc((size_t)BB * NN * C3 * 2);    // N-major qkv (pre-dw)
    __bf16* qk_cm  = (__bf16*)alloc((size_t)BB * 2 * CCH * NN * 2); // C-major q,k (post-dw)
    __bf16* vT     = (__bf16*)alloc((size_t)BB * NN * CCH * 2);   // N-major v (post-dw)
    __bf16* wqkvb  = (__bf16*)alloc((size_t)C3 * CCH * 2);
    __bf16* wprojb = (__bf16*)alloc((size_t)CCH * CCH * 2);
    __bf16* attnb  = (__bf16*)alloc((size_t)BB * CCH * CCH * 2);
    float*  S      = (float*)alloc((size_t)BB * CCH * CCH * 4);
    float*  rn     = (float*)alloc((size_t)BB * 2 * CCH * 4);

    // 0) Layout conversions: x -> bf16 NHWC; weights -> bf16
    transpose_x_kernel<<<dim3(NN / 32, CCH / 32, BB), 256, 0, stream>>>(x, xT);
    f32_to_bf16_kernel<<<dim3(C3 * CCH / 256), 256, 0, stream>>>(w_qkv, wqkvb, C3 * CCH);
    f32_to_bf16_kernel<<<dim3(CCH * CCH / 256), 256, 0, stream>>>(w_proj, wprojb, CCH * CCH);

    // 1) qkvT = (w_qkv @ x + b_qkv), bf16 N-major; K = 256
    gemm_wmma_kernel<0, true, false><<<dim3(NN / 32, C3 / 64, BB), 32, 0, stream>>>(
        wqkvb, 0, CCH, xT, (long long)NN * CCH, CCH, CCH, b_qkv,
        nullptr, 0, qkvT, (long long)NN * C3, C3);

    // 2) Depthwise 3x3 + bias: v -> N-major, q/k -> C-major (fused transpose)
    dwconv_v_kernel<<<dim3(NN, BB), 256, 0, stream>>>(qkvT, w_dw, b_dw, vT);
    dwconv_qk_t_kernel<<<dim3(NN / 32, 2 * CCH / 32, BB), 256, 0, stream>>>(
        qkvT, w_dw, b_dw, qk_cm);

    // 3) Inverse L2 norms of q and k rows (contiguous reductions)
    rnorm_kernel<<<dim3(2 * CCH, BB), 256, 0, stream>>>(qk_cm, rn);

    // 4) Raw logits S = Q @ K^T; contraction over N = 3136, all-b128 loads
    gemm_wmma_kernel<1, false, false><<<dim3(CCH / 32, CCH / 64, BB), 32, 0, stream>>>(
        qk_cm, (long long)2 * CCH * NN, NN,
        qk_cm + (size_t)CCH * NN, (long long)2 * CCH * NN, NN, NN,
        nullptr, nullptr, 0, S, (long long)CCH * CCH, CCH);

    // 5) Softmax with folded norms + temperature -> bf16 attn
    softmax_kernel<<<dim3(CCH, BB), 256, 0, stream>>>(S, rn, temp, attnb);

    // 6) out1T = attn @ V  (bf16 N-major; reuses dead xT buffer); K = 256
    __bf16* out1T = xT;
    gemm_wmma_kernel<0, false, false><<<dim3(NN / 32, CCH / 64, BB), 32, 0, stream>>>(
        attnb, (long long)CCH * CCH, CCH,
        vT, (long long)NN * CCH, CCH, CCH, nullptr,
        nullptr, 0, out1T, (long long)NN * CCH, CCH);

    // 7) out = w_proj @ out1 + b_proj + x  (f32 C-major, fused residual)
    gemm_wmma_kernel<1, true, true><<<dim3(NN / 32, CCH / 64, BB), 32, 0, stream>>>(
        wprojb, 0, CCH, out1T, (long long)NN * CCH, CCH, CCH, b_proj,
        x, (long long)CCH * NN, out, (long long)CCH * NN, NN);
}
